// Analysis_net_38972533244004
// MI455X (gfx1250) — 8-bit matrix operand order repaired and checked by reading, compiled, not run on hardware
//
#include <hip/hip_runtime.h>
#include <stdint.h>

typedef int v8i __attribute__((ext_vector_type(8)));

constexpr size_t AL(size_t v) { return (v + 255) & ~(size_t)255; }
constexpr size_t XQ_OFF  = 0;
constexpr size_t XQ_SZ   = AL((size_t)4 * 3 * 516 * 516);
constexpr size_t A1LO_OFF = XQ_OFF + XQ_SZ;
constexpr size_t A1_SZ   = AL((size_t)4 * 260 * 260 * 192);
constexpr size_t A1HI_OFF = A1LO_OFF + A1_SZ;
constexpr size_t A2LO_OFF = A1HI_OFF + A1_SZ;
constexpr size_t A2_SZ   = AL((size_t)4 * 132 * 132 * 192);
constexpr size_t A2HI_OFF = A2LO_OFF + A2_SZ;
constexpr size_t A3LO_OFF = A2HI_OFF + A2_SZ;
constexpr size_t A3_SZ   = AL((size_t)4 * 68 * 68 * 192);
constexpr size_t A3HI_OFF = A3LO_OFF + A3_SZ;
constexpr size_t ZSPAN   = A3HI_OFF + A3_SZ;
constexpr size_t WP1_OFF = ZSPAN;                 constexpr size_t WP1_SZ = AL((size_t)192 * 75);
constexpr size_t WP2_OFF = WP1_OFF + WP1_SZ;      constexpr size_t WP2_SZ = AL((size_t)192 * 4800);
constexpr size_t WP3_OFF = WP2_OFF + WP2_SZ;      constexpr size_t WP3_SZ = WP2_SZ;
constexpr size_t WP4_OFF = WP3_OFF + WP3_SZ;      constexpr size_t WP4_SZ = AL((size_t)320 * 4800);
constexpr size_t BI_OFF  = WP4_OFF + WP4_SZ;      constexpr size_t BI_SZ  = AL((size_t)896 * 4);
constexpr size_t WS_NEED = BI_OFF + BI_SZ;

#define LDT 80

__device__ __forceinline__ unsigned short act_quant(float acc, float mul, float addc,
                                                    float divc, float relu) {
  float y = acc * mul;
  float t = floorf((y + addc) / divc);
  float clp = rintf(255.0f / relu * 8388608.0f);
  t = fminf(fmaxf(t, 0.0f), clp);
  float scl = floorf((relu + 8.0f) / 16.0f);
  return (unsigned short)floorf((rintf(t * scl) + 262144.0f) / 524288.0f);
}

__global__ void k_quant_input(const float* __restrict__ x, uint8_t* __restrict__ xq) {
  int i = blockIdx.x * 256 + threadIdx.x;
  int wx = i & 511; int t = i >> 9; int hy = t & 511; t >>= 9;
  float v = fminf(fmaxf(rintf(x[i] * 256.0f), 0.0f), 255.0f);
  xq[((size_t)t * 516 + hy + 2) * 516 + wx + 2] = (uint8_t)v;
}

__global__ void k_pack_w1(const float* __restrict__ w, int8_t* __restrict__ d) {
  int i = blockIdx.x * 256 + threadIdx.x;
  if (i < 192 * 75) d[i] = (int8_t)rintf(w[i]);
}

__global__ void k_pack_w(const float* __restrict__ w, int8_t* __restrict__ d, int total) {
  int i = blockIdx.x * 256 + threadIdx.x;
  if (i >= total) return;
  int co = i / 4800, rem = i % 4800;
  int ci = rem / 25, tap = rem % 25;
  d[(size_t)co * 4800 + tap * 192 + ci] = (int8_t)rintf(w[i]);
}

__global__ void k_bias(const float* b1, const float* b2, const float* b3, const float* b4,
                       int* __restrict__ out) {
  int i = blockIdx.x * 256 + threadIdx.x;
  if (i < 192)       out[i] = (int)rintf(b1[i]);
  else if (i < 384)  out[i] = (int)rintf(b2[i - 192]);
  else if (i < 576)  out[i] = (int)rintf(b3[i - 384]);
  else if (i < 896)  out[i] = (int)rintf(b4[i - 576]);
}

__global__ void k_conv1(const uint8_t* __restrict__ xq, const int8_t* __restrict__ wq,
                        const int* __restrict__ bias, const float* __restrict__ muls,
                        const float* __restrict__ relus,
                        uint8_t* __restrict__ aLo, uint8_t* __restrict__ aHi) {
  int pix = blockIdx.x * 256 + threadIdx.x;
  int co = blockIdx.y, n = blockIdx.z;
  int ho = pix >> 8, wo = pix & 255;
  int acc = bias[co];
#pragma unroll
  for (int ci = 0; ci < 3; ++ci) {
    const uint8_t* p = xq + ((size_t)(n * 3 + ci) * 516 + 2 * ho) * 516 + 2 * wo;
    const int8_t* w = wq + (co * 3 + ci) * 25;
#pragma unroll
    for (int kh = 0; kh < 5; ++kh)
#pragma unroll
      for (int kw = 0; kw < 5; ++kw)
        acc += (int)w[kh * 5 + kw] * (int)p[kh * 516 + kw];
  }
  unsigned short u = act_quant((float)acc, muls[co], 524288.0f, 1048576.0f, relus[0]);
  size_t o = ((size_t)(n * 260 + ho + 2) * 260 + wo + 2) * 192 + co;
  aLo[o] = (uint8_t)(u & 255);
  aHi[o] = (uint8_t)(u >> 8);
}

__global__ void __launch_bounds__(256)
k_gemm(const uint8_t* __restrict__ inLo, const uint8_t* __restrict__ inHi,
       int Hp, int Hout, int Wout,
       const int8_t* __restrict__ wp, const int* __restrict__ bias,
       const float* __restrict__ muls, const float* __restrict__ relus, int lidx,
       float addc, float divc,
       uint8_t* __restrict__ outLo, uint8_t* __restrict__ outHi, int HpOut,
       float* __restrict__ finalOut, int Cout) {
  __shared__ __align__(16) uint8_t sA[2][64 * LDT];
  __shared__ __align__(16) uint8_t sBlo[2][64 * LDT];
  __shared__ __align__(16) uint8_t sBhi[2][64 * LDT];

  const int tid = threadIdx.x;
  const int m0 = blockIdx.y * 64, n0 = blockIdx.x * 64;

  const int jcol = tid >> 2, ch = tid & 3;
  const int colN = n0 + jcol;
  const int wo = colN % Wout;
  int tq = colN / Wout;
  const int ho = tq % Hout;
  const int nimg = tq / Hout;
  const unsigned aOff0 = (unsigned)((m0 + jcol) * 4800 + ch * 16);
  const unsigned ldsSlot = (unsigned)(jcol * LDT + ch * 16);
  const uint64_t wpB = (uint64_t)wp;
  const uint64_t loB = (uint64_t)inLo;
  const uint64_t hiB = (uint64_t)inHi;

  auto issue = [&](int kb, int buf) {
    int tap = kb / 3, cb = kb - tap * 3;
    int kh = tap / 5, kw = tap - kh * 5;
    unsigned aoff = aOff0 + (unsigned)(kb * 64);
    unsigned boff = (unsigned)(((nimg * Hp + 2 * ho + kh) * Hp + 2 * wo + kw) * 192
                               + cb * 64 + ch * 16);
    unsigned la = (unsigned)(uintptr_t)&sA[buf][ldsSlot];
    unsigned ll = (unsigned)(uintptr_t)&sBlo[buf][ldsSlot];
    unsigned lh = (unsigned)(uintptr_t)&sBhi[buf][ldsSlot];
    asm volatile("global_load_async_to_lds_b128 %0, %1, %2"
                 :: "v"(la), "v"(aoff), "s"(wpB) : "memory");
    asm volatile("global_load_async_to_lds_b128 %0, %1, %2"
                 :: "v"(ll), "v"(boff), "s"(loB) : "memory");
    asm volatile("global_load_async_to_lds_b128 %0, %1, %2"
                 :: "v"(lh), "v"(boff), "s"(hiB) : "memory");
  };

  const int lane = tid & 31, wv = tid >> 5;
  const int mi = wv & 3, njb = (wv >> 2) * 2;
  const int r = lane & 15, h = lane >> 4;

  const v8i vzero = {0, 0, 0, 0, 0, 0, 0, 0};
  v8i accLo[2], accHi[2];
  accLo[0] = vzero; accLo[1] = vzero; accHi[0] = vzero; accHi[1] = vzero;

  issue(0, 0);
  for (int kb = 0; kb < 75; ++kb) {
    const int buf = kb & 1;
    asm volatile("s_wait_asynccnt 0x0" ::: "memory");
    __syncthreads();
    if (kb + 1 < 75) issue(kb + 1, buf ^ 1);

    const uint8_t* bA = sA[buf];
    const uint8_t* bL = sBlo[buf];
    const uint8_t* bH = sBhi[buf];
    v8i A;
    {
      int ar = (mi * 16 + r) * LDT + h * 16;
      int2 q0 = *(const int2*)&bA[ar];
      int2 q1 = *(const int2*)&bA[ar + 8];
      int2 q2 = *(const int2*)&bA[ar + 32];
      int2 q3 = *(const int2*)&bA[ar + 40];
      A[0] = q0.x; A[1] = q0.y; A[2] = q1.x; A[3] = q1.y;
      A[4] = q2.x; A[5] = q2.y; A[6] = q3.x; A[7] = q3.y;
    }
#pragma unroll
    for (int p = 0; p < 2; ++p) {
      int boff = ((njb + p) * 16 + r) * LDT + h * 16;
      int4 b0 = *(const int4*)&bL[boff];
      int4 b1 = *(const int4*)&bL[boff + 32];
      v8i B;
      B[0] = b0.x; B[1] = b0.y; B[2] = b0.z; B[3] = b0.w;
      B[4] = b1.x; B[5] = b1.y; B[6] = b1.z; B[7] = b1.w;
      accLo[p] = __builtin_amdgcn_wmma_i32_16x16x64_iu8(
          true, A, false, B, accLo[p], false, false);
      int4 c0 = *(const int4*)&bH[boff];
      int4 c1 = *(const int4*)&bH[boff + 32];
      v8i Bh;
      Bh[0] = c0.x; Bh[1] = c0.y; Bh[2] = c0.z; Bh[3] = c0.w;
      Bh[4] = c1.x; Bh[5] = c1.y; Bh[6] = c1.z; Bh[7] = c1.w;
      accHi[p] = __builtin_amdgcn_wmma_i32_16x16x64_iu8(
          true, A, false, Bh, accHi[p], false, false);
    }
  }

  float relu = finalOut ? 1.0f : relus[lidx];
#pragma unroll
  for (int p = 0; p < 2; ++p) {
    int col = n0 + (njb + p) * 16 + r;
    int wo2 = col % Wout; int t2 = col / Wout;
    int ho2 = t2 % Hout; int n2 = t2 / Hout;
#pragma unroll
    for (int v = 0; v < 8; ++v) {
      int m = m0 + mi * 16 + v + 8 * h;
      int acc = accLo[p][v] + (accHi[p][v] << 8) + bias[m];
      float y = (float)acc * muls[m];
      if (finalOut) {
        finalOut[((size_t)(n2 * Cout + m) * Hout + ho2) * Wout + wo2] =
            floorf((y + addc) / divc);
      } else {
        unsigned short u = act_quant((float)acc, muls[m], addc, divc, relu);
        size_t o = ((size_t)(n2 * HpOut + ho2 + 2) * HpOut + wo2 + 2) * 192 + m;
        outLo[o] = (uint8_t)(u & 255);
        outHi[o] = (uint8_t)(u >> 8);
      }
    }
  }
}

extern "C" void kernel_launch(void* const* d_in, const int* in_sizes, int n_in,
                              void* d_out, int out_size, void* d_ws, size_t ws_size,
                              hipStream_t stream) {
  (void)in_sizes; (void)n_in; (void)out_size;
  if (ws_size < WS_NEED) return;

  const float* x     = (const float*)d_in[0];
  const float* muls0 = (const float*)d_in[1];
  const float* muls1 = (const float*)d_in[2];
  const float* muls2 = (const float*)d_in[3];
  const float* muls3 = (const float*)d_in[4];
  const float* relus = (const float*)d_in[5];
  const float* w1 = (const float*)d_in[6];
  const float* b1 = (const float*)d_in[7];
  const float* w2 = (const float*)d_in[8];
  const float* b2 = (const float*)d_in[9];
  const float* w3 = (const float*)d_in[10];
  const float* b3 = (const float*)d_in[11];
  const float* w4 = (const float*)d_in[12];
  const float* b4 = (const float*)d_in[13];

  uint8_t* ws = (uint8_t*)d_ws;
  uint8_t* xq   = ws + XQ_OFF;
  uint8_t* a1lo = ws + A1LO_OFF; uint8_t* a1hi = ws + A1HI_OFF;
  uint8_t* a2lo = ws + A2LO_OFF; uint8_t* a2hi = ws + A2HI_OFF;
  uint8_t* a3lo = ws + A3LO_OFF; uint8_t* a3hi = ws + A3HI_OFF;
  int8_t* wp1 = (int8_t*)(ws + WP1_OFF);
  int8_t* wp2 = (int8_t*)(ws + WP2_OFF);
  int8_t* wp3 = (int8_t*)(ws + WP3_OFF);
  int8_t* wp4 = (int8_t*)(ws + WP4_OFF);
  int* bi = (int*)(ws + BI_OFF);

  hipMemsetAsync(ws, 0, ZSPAN, stream);

  k_pack_w1<<<(192 * 75 + 255) / 256, 256, 0, stream>>>(w1, wp1);
  k_pack_w<<<(192 * 4800 + 255) / 256, 256, 0, stream>>>(w2, wp2, 192 * 4800);
  k_pack_w<<<(192 * 4800 + 255) / 256, 256, 0, stream>>>(w3, wp3, 192 * 4800);
  k_pack_w<<<(320 * 4800 + 255) / 256, 256, 0, stream>>>(w4, wp4, 320 * 4800);
  k_bias<<<4, 256, 0, stream>>>(b1, b2, b3, b4, bi);
  k_quant_input<<<(4 * 3 * 512 * 512) / 256, 256, 0, stream>>>(x, xq);

  k_conv1<<<dim3(256, 192, 4), 256, 0, stream>>>(xq, wp1, bi, muls0, relus, a1lo, a1hi);

  k_gemm<<<dim3(65536 / 64, 192 / 64), 256, 0, stream>>>(
      a1lo, a1hi, 260, 128, 128, wp2, bi + 192, muls1, relus, 1,
      32768.0f, 65536.0f, a2lo, a2hi, 132, nullptr, 192);

  k_gemm<<<dim3(16384 / 64, 192 / 64), 256, 0, stream>>>(
      a2lo, a2hi, 132, 64, 64, wp3, bi + 384, muls2, relus, 2,
      32768.0f, 65536.0f, a3lo, a3hi, 68, nullptr, 192);

  k_gemm<<<dim3(4096 / 64, 320 / 64), 256, 0, stream>>>(
      a3lo, a3hi, 68, 32, 32, wp4, bi + 576, muls3, relus, 0,
      16384.0f, 32768.0f, nullptr, nullptr, 0, (float*)d_out, 320);
}
